// DuelingScorerEdgeQLayer_70669391888817
// MI455X (gfx1250) — matrix operand order repaired and checked by reading, not compiled, not run on hardware
//
#include <hip/hip_runtime.h>
#include <hip/hip_bf16.h>

typedef unsigned short u16_t;
typedef __attribute__((ext_vector_type(16))) __bf16 v16bf;
typedef __attribute__((ext_vector_type(8)))  float  v8f;

#define BATCH  8
#define D2     256     // 2*EMBED
#define EMBED  128
#define NEF    32
#define EPB    16      // edges per block
#define ROWS   128     // BATCH * EPB
#define A_STR  40      // 32 + 8 pad (bf16 elems) -> 80B row, bank-spread
#define SA_STR 136     // 128 + 8 pad (bf16 elems) -> 272B row, bank-spread
#define NMAX_S 10016   // scores LDS capacity for attention kernel

__device__ __forceinline__ u16_t f2bf(float x) {
  unsigned u = __float_as_uint(x);
  u += 0x7FFFu + ((u >> 16) & 1u);   // round-to-nearest-even
  return (u16_t)(u >> 16);
}

union FragU { uint4 q[2]; v16bf v; };

// A-fragment (16x32 bf16, M-tile rows [rowBase, rowBase+16)) from LDS row-major tile.
// ISA layout: lane<16 -> row=lane, K in {k0..k0+7} U {k0+16..k0+23};
//             lane>=16 -> row=lane-16, K in {k0+8..k0+15} U {k0+24..k0+31}.
__device__ __forceinline__ v16bf lds_a_frag(const u16_t* s, int rowBase, int strideE, int k0) {
  int lane = threadIdx.x & 31;
  int hi = lane >> 4, lo = lane & 15;
  const u16_t* p = s + (rowBase + lo) * strideE + k0 + hi * 8;
  FragU f;
  f.q[0] = *(const uint4*)(p);        // K: k0+hi*8 .. +7   (8 contiguous bf16)
  f.q[1] = *(const uint4*)(p + 16);   // K: k0+16+hi*8 .. +7
  return f.v;
}

// B-fragment (32x16 bf16) from global bf16 weights stored row-major [out][Kdim].
// B[k][n] = W[n0+n][k]. ISA layout: lane<16 -> col=lane, K in {k0..k0+7} U {k0+16..k0+23}; lane>=16 -> col=lane-16, K in {k0+8..k0+15} U {k0+24..k0+31}.
__device__ __forceinline__ v16bf gbl_b_frag(const u16_t* w, int Kdim, int n0, int k0) {
  int lane = threadIdx.x & 31;
  int hi = lane >> 4, lo = lane & 15;
  const u16_t* p = w + (size_t)(n0 + lo) * Kdim + k0 + hi * 8;
  FragU f;
  f.q[0] = *(const uint4*)(p);
  f.q[1] = *(const uint4*)(p + 16);
  return f.v;
}

__device__ __forceinline__ v8f wmma_bf16(v16bf a, v16bf b, v8f c) {
  // D = A(16x32) x B(32x16) + C, f32 accumulate
  return __builtin_amdgcn_wmma_f32_16x16x32_bf16(false, a, false, b, (short)0, c, false, false);
}

// CDNA5 async global->LDS copy: 32 bytes per lane (two b128 transfers), ASYNCcnt-tracked.
// the immediate offset is added to both the LDS and the global address
__device__ __forceinline__ void async_stage32B(const u16_t* src, u16_t* dst) {
  unsigned l0 = (unsigned)(size_t)(void*)dst;   // low 32 bits of flat shared addr == LDS offset
  asm volatile("global_load_async_to_lds_b128 %0, %1, off"
               :: "v"(l0), "v"(src) : "memory");
  asm volatile("global_load_async_to_lds_b128 %0, %1, off offset:16"
               :: "v"(l0), "v"(src) : "memory");
}

template <int N>
__device__ __forceinline__ void wait_asynccnt() {
  asm volatile("s_wait_asynccnt %0" :: "i"(N) : "memory");
}

// ---------------------------------------------------------------- prep kernels
__global__ void cvt_bf16_kernel(const float* __restrict__ src, u16_t* __restrict__ dst, int n) {
  int i = blockIdx.x * 256 + threadIdx.x;
  if (i < n) dst[i] = f2bf(src[i]);
}

// repack adv1_w[:, 128:512] as contiguous bf16 [128][384]
__global__ void pack_w1_kernel(const float* __restrict__ adv1_w, u16_t* __restrict__ dst) {
  int i = blockIdx.x * 256 + threadIdx.x;
  if (i < 128 * 384) {
    int o = i / 384, k = i - o * 384;
    dst[i] = f2bf(adv1_w[o * 512 + 128 + k]);
  }
}

__global__ void zero_kernel(float* p, int n) {
  int i = blockIdx.x * 256 + threadIdx.x;
  if (i < n) p[i] = 0.f;
}

// --------------------------------------------- per-batch attention / pool / val / hbias
__global__ __launch_bounds__(256) void attn_g_kernel(
    const float* __restrict__ emb, const float* __restrict__ attn_w, const float* __restrict__ attn_b,
    const float* __restrict__ sp_w, const float* __restrict__ sp_b,
    const float* __restrict__ val1_w, const float* __restrict__ val1_b,
    const float* __restrict__ val2_w, const float* __restrict__ val2_b,
    const float* __restrict__ adv1_w, const float* __restrict__ adv1_b,
    float* __restrict__ val_out, float* __restrict__ hbias_out, int N) {
  __shared__ float sc[NMAX_S];
  __shared__ float red[256];
  __shared__ float aw[D2];
  __shared__ float g[D2];
  __shared__ float rg[EMBED];
  __shared__ float v1[EMBED];
  int b = blockIdx.x, t = threadIdx.x;
  const float* E = emb + (size_t)b * N * D2;
  aw[t] = attn_w[t];
  __syncthreads();

  float lmax = -3.4e38f;
  for (int n = t; n < N; n += 256) {
    const float* row = E + (size_t)n * D2;
    float s = 0.f;
    for (int d = 0; d < D2; d += 4) {
      float4 r = *(const float4*)(row + d);
      s += r.x * aw[d] + r.y * aw[d + 1] + r.z * aw[d + 2] + r.w * aw[d + 3];
    }
    s += attn_b[0];
    sc[n] = s;
    lmax = fmaxf(lmax, s);
  }
  red[t] = lmax; __syncthreads();
  for (int o = 128; o > 0; o >>= 1) { if (t < o) red[t] = fmaxf(red[t], red[t + o]); __syncthreads(); }
  float m = red[0]; __syncthreads();

  float lsum = 0.f;
  for (int n = t; n < N; n += 256) { float e = __expf(sc[n] - m); sc[n] = e; lsum += e; }
  red[t] = lsum; __syncthreads();
  for (int o = 128; o > 0; o >>= 1) { if (t < o) red[t] += red[t + o]; __syncthreads(); }
  float inv = 1.f / red[0];
  __syncthreads();

  // g[d] = sum_n alpha_n * emb[b][n][d]  (thread t owns dim d=t; coalesced over threads)
  float acc = 0.f;
  for (int n = 0; n < N; ++n) acc += sc[n] * E[(size_t)n * D2 + t];
  g[t] = acc * inv;
  __syncthreads();

  if (t < EMBED) {                      // rg = relu(sp(g))
    float s = sp_b[t];
    for (int d = 0; d < D2; ++d) s += sp_w[t * D2 + d] * g[d];
    rg[t] = fmaxf(s, 0.f);
  }
  __syncthreads();
  if (t < EMBED) {                      // val hidden
    float s = val1_b[t];
    for (int d = 0; d < EMBED; ++d) s += val1_w[t * EMBED + d] * rg[d];
    v1[t] = fmaxf(s, 0.f);
  }
  if (t < EMBED) {                      // hbias = adv1_w[:, :128] @ rg + adv1_b
    float s = adv1_b[t];
    for (int k = 0; k < EMBED; ++k) s += adv1_w[t * 512 + k] * rg[k];
    hbias_out[b * EMBED + t] = s;
  }
  __syncthreads();
  if (t == 0) {
    float s = val2_b[0];
    for (int o = 0; o < EMBED; ++o) s += val2_w[o] * v1[o];
    val_out[b] = s;
  }
}

// --------------------------------------------------------------- WMMA edge kernel
__global__ __launch_bounds__(256) void edge_kernel(
    const u16_t* __restrict__ emb_bf, const u16_t* __restrict__ ef_bf,
    const int* __restrict__ edges,
    const u16_t* __restrict__ spw_bf, const float* __restrict__ sp_b,
    const u16_t* __restrict__ apw_bf, const float* __restrict__ ap_b,
    const u16_t* __restrict__ w1p_bf, const float* __restrict__ hbias,
    const float* __restrict__ adv2_w, const float* __restrict__ adv2_b,
    float* __restrict__ adv_out, float* __restrict__ adv_sum, int Nn, int Etot) {
  __shared__ __attribute__((aligned(16))) u16_t Achunk[2][ROWS * A_STR];  // 2 x 10.0 KB (double buffer)
  __shared__ __attribute__((aligned(16))) u16_t SAbuf[ROWS * SA_STR];     // 34.0 KB
  __shared__ int   idxs[2][EPB];
  __shared__ float hb[BATCH * EMBED];
  __shared__ float w2[EMBED];
  __shared__ float advred[ROWS];

  int t = threadIdx.x;
  int wave = t >> 5;
  int n0 = wave * 16;                 // this wave's output-feature tile
  int e0 = blockIdx.x * EPB;

  if (t < EPB) {
    int ee = e0 + t;
    idxs[0][t] = (ee < Etot) ? edges[ee * 2 + 0] : 0;
    idxs[1][t] = (ee < Etot) ? edges[ee * 2 + 1] : 0;
  }
  if (t < EMBED) { w2[t] = adv2_w[t]; advred[t] = 0.f; }
  for (int i = t; i < BATCH * EMBED; i += 256) hb[i] = hbias[i];
  __syncthreads();

  // this thread's staging assignment: 32 bytes of one activation row per k-step
  int srow = t >> 1, shf = t & 1;
  int sb = srow >> 4, sel = srow & 15;
  u16_t* sdst0 = &Achunk[0][0] + srow * A_STR + shf * 16;
  u16_t* sdst1 = &Achunk[1][0] + srow * A_STR + shf * 16;

  v8f acc_h[8];                       // h accumulators: 8 M-tiles (one per batch)
  for (int m2 = 0; m2 < 8; ++m2)
    for (int r = 0; r < 8; ++r) acc_h[m2][r] = 0.f;

  // parts: 0 = mu_i (K=256), 1 = mu_j (K=256), 2 = a (K=32)
  for (int part = 0; part < 3; ++part) {
    int Kdim = (part == 2) ? NEF : D2;
    const u16_t* W = (part == 2) ? apw_bf : spw_bf;
    const float* bias = (part == 2) ? ap_b : sp_b;
    int nk = Kdim / 32;

    // per-thread gather source base (row-major bf16, this thread's 16-elem half)
    const u16_t* srcbase;
    if (part == 2) {
      int ee = min(e0 + sel, Etot - 1);
      srcbase = ef_bf + ((size_t)sb * Etot + ee) * NEF + shf * 16;
    } else {
      int node = idxs[part][sel];
      srcbase = emb_bf + ((size_t)sb * Nn + node) * D2 + shf * 16;
    }

    v8f accp[8];
    for (int m2 = 0; m2 < 8; ++m2)
      for (int r = 0; r < 8; ++r) accp[m2][r] = 0.f;

    // double-buffered async pipeline: stage k+1 while computing k
    async_stage32B(srcbase, sdst0);                       // prologue -> buf0
    for (int ks = 0; ks < nk; ++ks) {
      int cur = ks & 1;
      if (ks + 1 < nk) {
        async_stage32B(srcbase + (ks + 1) * 32, cur ? sdst0 : sdst1);
        wait_asynccnt<2>();   // in-order returns: oldest 2 (current buffer) done
      } else {
        wait_asynccnt<0>();
      }
      __syncthreads();        // barrier A: current buffer visible to all waves
      v16bf bfrag = gbl_b_frag(W, Kdim, n0, ks * 32);
      const u16_t* abuf = &Achunk[cur][0];
      for (int m2 = 0; m2 < 8; ++m2) {
        v16bf afrag = lds_a_frag(abuf, m2 * 16, A_STR, 0);
        accp[m2] = wmma_bf16(afrag, bfrag, accp[m2]);
      }
      __syncthreads();        // barrier B: all reads done; buffer may be overwritten
    }

    { // bias + relu -> bf16 SA chunk in LDS (C layout: row = r + 8*hi, col = lo)
      int lane = t & 31, hi = lane >> 4, lo = lane & 15;
      float bn = bias[n0 + lo];
      for (int m2 = 0; m2 < 8; ++m2) {
        int rowBase = m2 * 16 + hi * 8;
        for (int r = 0; r < 8; ++r) {
          float v = fmaxf(accp[m2][r] + bn, 0.f);
          SAbuf[(rowBase + r) * SA_STR + n0 + lo] = f2bf(v);
        }
      }
    }
    __syncthreads();

    // stage 2: accumulate h += SA_chunk(128x128) @ adv1_w-slice(128x128)
    for (int k2 = 0; k2 < EMBED; k2 += 32) {
      v16bf bfrag = gbl_b_frag(w1p_bf, 384, n0, part * EMBED + k2);
      for (int m2 = 0; m2 < 8; ++m2) {
        v16bf afrag = lds_a_frag(SAbuf, m2 * 16, SA_STR, k2);
        acc_h[m2] = wmma_bf16(afrag, bfrag, acc_h[m2]);
      }
    }
    __syncthreads();
  }

  { // adv partials: relu(h + hbias[batch]) . adv2_w, reduced over this wave's 16 cols
    int lane = t & 31, hi = lane >> 4, lo = lane & 15;
    for (int m2 = 0; m2 < 8; ++m2) {          // m2 == batch
      float hbn = hb[m2 * EMBED + n0 + lo];
      float p8[8];
      for (int r = 0; r < 8; ++r) {
        float v = fmaxf(acc_h[m2][r] + hbn, 0.f);
        p8[r] = v * w2[n0 + lo];
      }
      for (int off = 1; off < 16; off <<= 1)
        for (int r = 0; r < 8; ++r)
          p8[r] += __shfl_xor(p8[r], off, 32);
      if (lo == 0)
        for (int r = 0; r < 8; ++r)
          atomicAdd(&advred[m2 * 16 + hi * 8 + r], p8[r]);   // ds_add_f32
    }
  }
  __syncthreads();

  float advv = 0.f;
  if (t < ROWS) {
    advv = advred[t] + adv2_b[0];
    int b = t >> 4, el = t & 15;
    if (e0 + el < Etot) adv_out[(size_t)b * Etot + e0 + el] = advv;
  }
  __syncthreads();
  if (t < ROWS) advred[t] = (e0 + (t & 15) < Etot) ? advv : 0.f;
  __syncthreads();
  if (t < BATCH) {
    float s = 0.f;
    for (int el = 0; el < EPB; ++el) s += advred[t * EPB + el];
    atomicAdd(&adv_sum[t], s);
  }
}

// ---------------------------------------------------------------- finalize
__global__ void finalize_kernel(const float* __restrict__ adv, const float* __restrict__ val,
                                const float* __restrict__ advsum, float* __restrict__ q, int Etot) {
  int i = blockIdx.x * 256 + threadIdx.x;
  if (i < BATCH * Etot) {
    int b = i / Etot;
    q[i] = val[b] + adv[i] - advsum[b] * (1.f / (float)Etot);
  }
}

extern "C" void kernel_launch(void* const* d_in, const int* in_sizes, int n_in,
                              void* d_out, int out_size, void* d_ws, size_t ws_size,
                              hipStream_t stream) {
  const float* emb    = (const float*)d_in[0];
  const float* ef     = (const float*)d_in[1];
  const int*   edges  = (const int*)d_in[2];
  const float* attn_w = (const float*)d_in[3];
  const float* attn_b = (const float*)d_in[4];
  const float* sp_w   = (const float*)d_in[5];
  const float* sp_b   = (const float*)d_in[6];
  const float* ap_w   = (const float*)d_in[7];
  const float* ap_b   = (const float*)d_in[8];
  const float* adv1_w = (const float*)d_in[9];
  const float* adv1_b = (const float*)d_in[10];
  const float* adv2_w = (const float*)d_in[11];
  const float* adv2_b = (const float*)d_in[12];
  const float* val1_w = (const float*)d_in[13];
  const float* val1_b = (const float*)d_in[14];
  const float* val2_w = (const float*)d_in[15];
  const float* val2_b = (const float*)d_in[16];
  (void)n_in; (void)out_size; (void)ws_size;

  int N = in_sizes[0] / (BATCH * D2);
  int E = in_sizes[2] / 2;

  char* ws = (char*)d_ws;
  size_t off = 0;
  auto walloc = [&](size_t bytes) -> void* {
    void* p = (void*)(ws + off);
    off = (off + bytes + 255) & ~(size_t)255;
    return p;
  };
  u16_t* emb_bf = (u16_t*)walloc((size_t)BATCH * N * D2 * 2);
  u16_t* ef_bf  = (u16_t*)walloc((size_t)BATCH * E * NEF * 2);
  u16_t* spw_bf = (u16_t*)walloc((size_t)EMBED * D2 * 2);
  u16_t* apw_bf = (u16_t*)walloc((size_t)EMBED * NEF * 2);
  u16_t* w1p_bf = (u16_t*)walloc((size_t)EMBED * 384 * 2);
  float* hbias  = (float*)walloc((size_t)BATCH * EMBED * 4);
  float* val_b  = (float*)walloc((size_t)BATCH * 4);
  float* adv    = (float*)walloc((size_t)BATCH * E * 4);
  float* advsum = (float*)walloc((size_t)BATCH * 4);

  int nEmb = BATCH * N * D2;
  int nEf  = BATCH * E * NEF;
  cvt_bf16_kernel<<<(nEmb + 255) / 256, 256, 0, stream>>>(emb, emb_bf, nEmb);
  cvt_bf16_kernel<<<(nEf + 255) / 256, 256, 0, stream>>>(ef, ef_bf, nEf);
  cvt_bf16_kernel<<<(EMBED * D2 + 255) / 256, 256, 0, stream>>>(sp_w, spw_bf, EMBED * D2);
  cvt_bf16_kernel<<<(EMBED * NEF + 255) / 256, 256, 0, stream>>>(ap_w, apw_bf, EMBED * NEF);
  pack_w1_kernel<<<(EMBED * 384 + 255) / 256, 256, 0, stream>>>(adv1_w, w1p_bf);
  zero_kernel<<<1, 256, 0, stream>>>(advsum, BATCH);

  attn_g_kernel<<<BATCH, 256, 0, stream>>>(emb, attn_w, attn_b, sp_w, sp_b,
                                           val1_w, val1_b, val2_w, val2_b,
                                           adv1_w, adv1_b, val_b, hbias, N);

  edge_kernel<<<(E + EPB - 1) / EPB, 256, 0, stream>>>(
      emb_bf, ef_bf, edges, spw_bf, sp_b, apw_bf, ap_b,
      w1p_bf, hbias, adv2_w, adv2_b, adv, advsum, N, E);

  finalize_kernel<<<(BATCH * E + 255) / 256, 256, 0, stream>>>(adv, val_b, advsum, (float*)d_out, E);
}
